// Block_49435073577068
// MI455X (gfx1250) — matrix operand order repaired and checked by reading, not compiled, not run on hardware
//
#include <hip/hip_runtime.h>

// ---------------------------------------------------------------------------
// AFNO block for MI455X (gfx1250, wave32, WMMA, async-LDS).
// All dense math (DFT-as-GEMM, spectral block-MLP, token MLP) runs on
// v_wmma_f32_16x16x32_bf16 (fp32 accumulate). All intermediates stored bf16:
// halves HBM traffic and keeps the 135MB spectral working set inside the
// 192MB L2. Complex FFT stages use a fused dual-pair GEMM (A1*B1 +/- A2*B2)
// so no beta read-modify-write passes are needed.
// ---------------------------------------------------------------------------

typedef __attribute__((ext_vector_type(16))) __bf16 v16bf;
typedef __attribute__((ext_vector_type(8)))  float  v8f;
typedef __attribute__((ext_vector_type(4)))  int    v4i;

#define B_   2
#define H_   256
#define W_   256
#define C_   256
#define KW_  129
#define NTOK (B_*H_*W_)      // 131072
#define NPOS (B_*H_*KW_)     // 66048

// workspace offsets (bytes); all bf16 unless noted
#define OFF_HLN1 0ULL            // 67108864  (bf16 LN1 out)
#define OFF_XR   67108864ULL     // 33816576
#define OFF_XI   100925440ULL    // 33816576
#define OFF_YR   134742016ULL    // 33816576
#define OFF_YI   168558592ULL    // 33816576
#define OFF_IRF  202375168ULL    // 67108864  (bf16 irfft out)
#define OFF_OUT1 269484032ULL    // 134217728 (f32 residual out1)
#define OFF_MATS 403701760ULL    // 788480    (bf16 DFT matrices)
#define OFF_T    404490240ULL    // 67108864  (bf16 LN2 out)
#define OFF_U    0ULL            // 268435456 (bf16 MLP hidden; overlays freed regions)
// DFT matrix offsets within OFF_MATS (bytes)
#define MOFF_FWC  0ULL
#define MOFF_FWS  66048ULL
#define MOFF_FHC  132096ULL
#define MOFF_FHS  263168ULL
#define MOFF_IFHC 394240ULL
#define MOFF_IFHS 525312ULL
#define MOFF_IFWC 656384ULL
#define MOFF_IFWS 722432ULL

#if defined(__gfx1250__) && __has_builtin(__builtin_amdgcn_global_load_async_to_lds_b128) && __has_builtin(__builtin_amdgcn_s_wait_asynccnt)
#define GFX1250_ASYNC 1
#else
#define GFX1250_ASYNC 0
#endif

__device__ __forceinline__ unsigned short f2bf(float f) {
  union { float f; unsigned u; } v; v.f = f;
  unsigned r = v.u + 0x7FFFu + ((v.u >> 16) & 1u);
  return (unsigned short)(r >> 16);
}
__device__ __forceinline__ float bf2f(unsigned short b) {
  union { unsigned u; float f; } v; v.u = ((unsigned)b) << 16; return v.f;
}

union Frag16 { v16bf v; unsigned short s[16]; uint4 q[2]; };

__device__ __forceinline__ v8f wmma_bf16(v16bf a, v16bf b, v8f c) {
  return __builtin_amdgcn_wmma_f32_16x16x32_bf16(false, a, false, b, (short)0, c,
                                                 false, false);
}

#if GFX1250_ASYNC
typedef __attribute__((address_space(3))) v4i* lds_v4i_t;
__device__ __forceinline__ void async_g2l_b128(const void* g, void* l) {
  // LDS dest via inttoptr: flat-address low 32 bits == LDS byte offset (ISA 10.2)
  __builtin_amdgcn_global_load_async_to_lds_b128(
      (v4i*)g, (lds_v4i_t)(unsigned)(unsigned long long)l, 0, 0);
}
#endif

// ------------------------------- tiled GEMM --------------------------------
#define BM 128
#define BN 128
#define BK 32
#define APAD 40
#define BPAD 40

// A-frag (16x32 bf16): lane l -> row l%16; g=l/16; k = g*16 + e.
__device__ __forceinline__ v16bf lds_a(const unsigned short* As, int rowBase, int lane) {
  int row = rowBase + (lane & 15), g = lane >> 4;
  Frag16 u;
  u.q[0] = *(const uint4*)(As + row * APAD + g * 16);
  u.q[1] = *(const uint4*)(As + row * APAD + g * 16 + 8);
  return u.v;
}
// B-frag (32x16 bf16): lane l -> col l%16; k = (l/16)*16 + e. Bs stored [n][k].
__device__ __forceinline__ v16bf lds_b(const unsigned short* Bs, int colBase, int lane) {
  int col = colBase + (lane & 15), g = lane >> 4;
  const uint4* p = (const uint4*)(Bs + col * BPAD + g * 16);
  Frag16 u; u.q[0] = p[0]; u.q[1] = p[1];
  return u.v;
}

// Stage a 128x32 bf16 A tile. Uses CDNA5 async global->LDS copies when the
// tile is fully in-bounds and 16B-aligned; returns true if async was used.
__device__ __forceinline__ bool stage_a_tile(unsigned short* As,
    const unsigned short* __restrict__ Ab, int lda, int blockM, int kk,
    int M, int K, int tid) {
#if GFX1250_ASYNC
  if ((blockM + BM <= M) && (kk + BK <= K) && ((lda & 7) == 0)) {
#pragma unroll
    for (int c = 0; c < 2; ++c) {
      int idx = c * 256 + tid;              // 512 chunks of 8 elements
      int m = idx >> 2, kc = idx & 3;
      async_g2l_b128(Ab + (long long)(blockM + m) * lda + kk + kc * 8,
                     As + m * APAD + kc * 8);
    }
    return true;
  }
#endif
#pragma unroll
  for (int r = 0; r < 16; ++r) {
    int idx = r * 256 + tid;
    int m = idx >> 5, k = idx & 31;
    int gm = blockM + m, gk = kk + k;
    As[m * APAD + k] = (gm < M && gk < K)
        ? Ab[(long long)gm * lda + gk] : (unsigned short)0;
  }
  return false;
}

// Stage a 32x128 B tile transposed into LDS [n][k]. bf16 source: vector fast
// path (2x uint4 loads/thread) when fully in-bounds; f32 source: scalar+cvt.
__device__ __forceinline__ void stage_b_tile(unsigned short* Bs,
    const unsigned short* __restrict__ Bb, int ldb, int blockN, int kk,
    int N, int K, unsigned xm, int tid) {
  if ((kk + BK <= K) && (blockN + BN <= N) && ((ldb & 7) == 0)) {
#pragma unroll
    for (int c = 0; c < 2; ++c) {
      int idx = c * 256 + tid;              // 512 chunks of 8 elements
      int k = idx >> 4, nc = idx & 15;
      uint4 q = *(const uint4*)(Bb + (long long)(kk + k) * ldb + blockN + nc * 8);
      const unsigned short* e = (const unsigned short*)&q;
#pragma unroll
      for (int j = 0; j < 8; ++j)
        Bs[(nc * 8 + j) * BPAD + k] = (unsigned short)(e[j] ^ xm);
    }
    return;
  }
#pragma unroll
  for (int r = 0; r < 16; ++r) {
    int idx = r * 256 + tid;
    int k = idx >> 7, n = idx & 127;
    int gk = kk + k, gn = blockN + n;
    unsigned short b = 0;
    if (gk < K && gn < N)
      b = (unsigned short)(Bb[(long long)gk * ldb + gn] ^ xm);
    Bs[n * BPAD + k] = b;
  }
}
__device__ __forceinline__ void stage_b_tile(unsigned short* Bs,
    const float* __restrict__ Bb, int ldb, int blockN, int kk,
    int N, int K, unsigned xm, int tid) {
#pragma unroll
  for (int r = 0; r < 16; ++r) {
    int idx = r * 256 + tid;
    int k = idx >> 7, n = idx & 127;
    int gk = kk + k, gn = blockN + n;
    unsigned short b = 0;
    if (gk < K && gn < N)
      b = (unsigned short)(f2bf(Bb[(long long)gk * ldb + gn]) ^ xm);
    Bs[n * BPAD + k] = b;
  }
}

// C = A1*B1 + (B2 sign-folded) A2*B2.   A always bf16 (bits). TB: bf16|f32.
// mode 0: store bf16      mode 1: bf16(gelu(acc+bias))      mode 2: f32 acc+bias+res
template <typename TB>
__global__ __launch_bounds__(256) void gemm_dual_kernel(
    const unsigned short* __restrict__ A1, const unsigned short* __restrict__ A2,
    int lda, long long sA,
    const TB* __restrict__ B1, const TB* __restrict__ B2,
    int ldb, long long sB, unsigned negB2,
    void* __restrict__ Cout, int ldc, long long sC,
    int M, int N, int K, int mode,
    const float* __restrict__ bias, const float* __restrict__ res) {
  __shared__ __attribute__((aligned(16))) unsigned short As1[BM * APAD];
  __shared__ __attribute__((aligned(16))) unsigned short As2[BM * APAD];
  __shared__ __attribute__((aligned(16))) unsigned short Bs1[BN * BPAD];
  __shared__ __attribute__((aligned(16))) unsigned short Bs2[BN * BPAD];
  const bool dual = (A2 != nullptr);
  const int z = blockIdx.z;
  const unsigned short* A1b = A1 + sA * z;
  const unsigned short* A2b = dual ? A2 + sA * z : nullptr;
  const TB* B1b = B1 + sB * z;
  const TB* B2b = dual ? B2 + sB * z : nullptr;
  const int blockM = blockIdx.y * BM, blockN = blockIdx.x * BN;
  const int tid = threadIdx.x, lane = tid & 31, wid = tid >> 5;
  const int waveM = (wid & 3) * 32, waveN = (wid >> 2) * 64;
  v8f acc[2][4] = {};
  for (int kk = 0; kk < K; kk += BK) {
    bool aa = stage_a_tile(As1, A1b, lda, blockM, kk, M, K, tid);
    if (dual) aa |= stage_a_tile(As2, A2b, lda, blockM, kk, M, K, tid);
    stage_b_tile(Bs1, B1b, ldb, blockN, kk, N, K, 0u, tid);
    if (dual) stage_b_tile(Bs2, B2b, ldb, blockN, kk, N, K, negB2, tid);
#if GFX1250_ASYNC
    if (aa) __builtin_amdgcn_s_wait_asynccnt(0);
#else
    (void)aa;
#endif
    __syncthreads();
    {
      v16bf af[2], bfr[4];
#pragma unroll
      for (int i = 0; i < 2; ++i) af[i] = lds_a(As1, waveM + i * 16, lane);
#pragma unroll
      for (int j = 0; j < 4; ++j) bfr[j] = lds_b(Bs1, waveN + j * 16, lane);
#pragma unroll
      for (int i = 0; i < 2; ++i)
#pragma unroll
        for (int j = 0; j < 4; ++j)
          acc[i][j] = wmma_bf16(af[i], bfr[j], acc[i][j]);
    }
    if (dual) {
      v16bf af[2], bfr[4];
#pragma unroll
      for (int i = 0; i < 2; ++i) af[i] = lds_a(As2, waveM + i * 16, lane);
#pragma unroll
      for (int j = 0; j < 4; ++j) bfr[j] = lds_b(Bs2, waveN + j * 16, lane);
#pragma unroll
      for (int i = 0; i < 2; ++i)
#pragma unroll
        for (int j = 0; j < 4; ++j)
          acc[i][j] = wmma_bf16(af[i], bfr[j], acc[i][j]);
    }
    __syncthreads();
  }
  const int g = lane >> 4, lm = lane & 15;
#pragma unroll
  for (int i = 0; i < 2; ++i)
#pragma unroll
    for (int j = 0; j < 4; ++j)
#pragma unroll
      for (int r = 0; r < 8; ++r) {
        int m = blockM + waveM + i * 16 + g * 8 + r;
        int n = blockN + waveN + j * 16 + lm;
        if (m < M && n < N) {
          float v = acc[i][j][r];
          long long off = (long long)m * ldc + n;
          if (mode == 0) {
            ((unsigned short*)Cout + sC * z)[off] = f2bf(v);
          } else if (mode == 1) {
            v += bias[n];
            v = 0.5f * v * (1.f + erff(v * 0.70710678118654752f));
            ((unsigned short*)Cout)[off] = f2bf(v);
          } else {
            v += bias[n] + res[off];
            ((float*)Cout)[off] = v;
          }
        }
      }
}

// ------------------------------- DFT matrices ------------------------------
__global__ __launch_bounds__(256) void build_mats_kernel(
    unsigned short* FwC, unsigned short* FwS, unsigned short* FhC,
    unsigned short* FhS, unsigned short* IFhC, unsigned short* IFhS,
    unsigned short* IFwC, unsigned short* IFwS) {
  int idx = blockIdx.x * 256 + threadIdx.x;   // 65536 entries
  int a = idx >> 8, b = idx & 255;
  float th = 6.28318530717958647692f * (float)((a * b) & 255) * (1.f / 256.f);
  float cv = cosf(th) * (1.f / 16.f);
  float sv = sinf(th) * (1.f / 16.f);
  FhC[idx] = f2bf(cv);  FhS[idx] = f2bf(-sv);   // forward: e^{-i t}
  IFhC[idx] = f2bf(cv); IFhS[idx] = f2bf(sv);   // inverse: e^{+i t}
  if (a < 129) { FwC[a * 256 + b] = f2bf(cv); FwS[a * 256 + b] = f2bf(-sv); }
  if (b < 129) {                                // irfft Hermitian weights
    float alpha = (b == 0 || b == 128) ? 1.f : 2.f;
    IFwC[a * 129 + b] = f2bf(alpha * cv);
    IFwS[a * 129 + b] = f2bf(-alpha * sv);
  }
}

// ------------------------------- LayerNorms --------------------------------
__global__ __launch_bounds__(256) void ln_kernel(const float* __restrict__ x,
    const float* __restrict__ gam, const float* __restrict__ bet,
    unsigned short* __restrict__ out) {
  __shared__ float s1[256], s2[256];
  const long long base = (long long)blockIdx.x * 256;
  const int c = threadIdx.x;
  float v = x[base + c];
  s1[c] = v; s2[c] = v * v;
  __syncthreads();
  for (int off = 128; off > 0; off >>= 1) {
    if (c < off) { s1[c] += s1[c + off]; s2[c] += s2[c + off]; }
    __syncthreads();
  }
  float mu = s1[0] * (1.f / 256.f);
  float var = s2[0] * (1.f / 256.f) - mu * mu;
  float rr = rsqrtf(var + 1e-5f);
  out[base + c] = f2bf((v - mu) * rr * gam[c] + bet[c]);
}

// out1 = irfft + ln1_out + x (f32, kept for final residual); t = bf16(LN2(out1))
__global__ __launch_bounds__(256) void add_ln2_kernel(
    const unsigned short* __restrict__ irf, const unsigned short* __restrict__ hln1,
    const float* __restrict__ x, const float* __restrict__ gam,
    const float* __restrict__ bet, float* __restrict__ out1,
    unsigned short* __restrict__ t) {
  __shared__ float s1[256], s2[256];
  const long long base = (long long)blockIdx.x * 256;
  const int c = threadIdx.x;
  float v = bf2f(irf[base + c]) + bf2f(hln1[base + c]) + x[base + c];
  s1[c] = v; s2[c] = v * v;
  __syncthreads();
  for (int off = 128; off > 0; off >>= 1) {
    if (c < off) { s1[c] += s1[c + off]; s2[c] += s2[c + off]; }
    __syncthreads();
  }
  float mu = s1[0] * (1.f / 256.f);
  float var = s2[0] * (1.f / 256.f) - mu * mu;
  float rr = rsqrtf(var + 1e-5f);
  out1[base + c] = v;
  t[base + c] = f2bf((v - mu) * rr * gam[c] + bet[c]);
}

// --------------------- spectral block-diagonal complex MLP -----------------
// 8 waves/WG, wave = one 32-channel block, WG = 16 frequency positions.
__global__ __launch_bounds__(256) void afno_filter_kernel(
    const unsigned short* __restrict__ Yr, const unsigned short* __restrict__ Yi,
    const float* __restrict__ w1, const float* __restrict__ b1,
    const float* __restrict__ w2, const float* __restrict__ b2,
    unsigned short* __restrict__ Or_, unsigned short* __restrict__ Oi_) {
  __shared__ float xfer[8][16][32];
  const int lane = threadIdx.x & 31, blk = threadIdx.x >> 5;
  const int g = lane >> 4, lm = lane & 15;
  const int posBase = blockIdx.x * 16;

  // weight frags in WMMA B layout: col = h*16+lm, k = (e&8)*2 + g*8 + (e&7)
  Frag16 W1r[2], W1iP[2], W1iN[2], W2r[2], W2iP[2], W2iN[2];
#pragma unroll
  for (int h = 0; h < 2; ++h) {
    int o = h * 16 + lm;
#pragma unroll
    for (int e = 0; e < 16; ++e) {
      int i = (e & 8) * 2 + g * 8 + (e & 7);
      float a1r = w1[((0 * 8 + blk) * 32 + i) * 32 + o];
      float a1i = w1[((1 * 8 + blk) * 32 + i) * 32 + o];
      float a2r = w2[((0 * 8 + blk) * 32 + i) * 32 + o];
      float a2i = w2[((1 * 8 + blk) * 32 + i) * 32 + o];
      W1r[h].s[e] = f2bf(a1r);  W1iP[h].s[e] = f2bf(a1i);  W1iN[h].s[e] = f2bf(-a1i);
      W2r[h].s[e] = f2bf(a2r);  W2iP[h].s[e] = f2bf(a2i);  W2iN[h].s[e] = f2bf(-a2i);
    }
  }

  // A frags: direct 16B loads of bf16 spectra (no conversion)
  const int pos = posBase + lm;
  const unsigned short* pr = Yr + (long long)pos * 256 + blk * 32;
  const unsigned short* pi = Yi + (long long)pos * 256 + blk * 32;
  Frag16 ar, ai;
  ar.q[0] = *(const uint4*)(pr + g * 8);
  ar.q[1] = *(const uint4*)(pr + 16 + g * 8);
  ai.q[0] = *(const uint4*)(pi + g * 8);
  ai.q[1] = *(const uint4*)(pi + 16 + g * 8);

  // layer 1: o1r = relu(Xr*W1r - Xi*W1i + b1r), o1i = relu(Xi*W1r + Xr*W1i + b1i)
  v8f o1r[2], o1i[2];
#pragma unroll
  for (int h = 0; h < 2; ++h) {
    float br = b1[(0 * 8 + blk) * 32 + h * 16 + lm];
    float bi = b1[(1 * 8 + blk) * 32 + h * 16 + lm];
    v8f aR, aI;
#pragma unroll
    for (int r = 0; r < 8; ++r) { aR[r] = br; aI[r] = bi; }
    aR = wmma_bf16(ar.v, W1r[h].v, aR);
    aR = wmma_bf16(ai.v, W1iN[h].v, aR);
    aI = wmma_bf16(ai.v, W1r[h].v, aI);
    aI = wmma_bf16(ar.v, W1iP[h].v, aI);
#pragma unroll
    for (int r = 0; r < 8; ++r) { aR[r] = fmaxf(aR[r], 0.f); aI[r] = fmaxf(aI[r], 0.f); }
    o1r[h] = aR; o1i[h] = aI;
  }

  // D-layout -> A-layout via per-wave LDS tile
  Frag16 fr, fi;
#pragma unroll
  for (int h = 0; h < 2; ++h)
#pragma unroll
    for (int r = 0; r < 8; ++r) xfer[blk][g * 8 + r][h * 16 + lm] = o1r[h][r];
  __syncthreads();
#pragma unroll
  for (int e = 0; e < 8; ++e) {
    fr.s[e] = f2bf(xfer[blk][lm][g * 8 + e]);
    fr.s[8 + e] = f2bf(xfer[blk][lm][16 + g * 8 + e]);
  }
  __syncthreads();
#pragma unroll
  for (int h = 0; h < 2; ++h)
#pragma unroll
    for (int r = 0; r < 8; ++r) xfer[blk][g * 8 + r][h * 16 + lm] = o1i[h][r];
  __syncthreads();
#pragma unroll
  for (int e = 0; e < 8; ++e) {
    fi.s[e] = f2bf(xfer[blk][lm][g * 8 + e]);
    fi.s[8 + e] = f2bf(xfer[blk][lm][16 + g * 8 + e]);
  }
  __syncthreads();

  // layer 2 + softshrink + bf16 store (filtered spectra overwrite X planes)
#pragma unroll
  for (int h = 0; h < 2; ++h) {
    float br = b2[(0 * 8 + blk) * 32 + h * 16 + lm];
    float bi = b2[(1 * 8 + blk) * 32 + h * 16 + lm];
    v8f aR, aI;
#pragma unroll
    for (int r = 0; r < 8; ++r) { aR[r] = br; aI[r] = bi; }
    aR = wmma_bf16(fr.v, W2r[h].v, aR);
    aR = wmma_bf16(fi.v, W2iN[h].v, aR);
    aI = wmma_bf16(fi.v, W2r[h].v, aI);
    aI = wmma_bf16(fr.v, W2iP[h].v, aI);
#pragma unroll
    for (int r = 0; r < 8; ++r) {
      int p = posBase + g * 8 + r;
      int c = blk * 32 + h * 16 + lm;
      float vr = aR[r], vi = aI[r];
      float mr = fabsf(vr) - 0.01f, mi = fabsf(vi) - 0.01f;
      Or_[(long long)p * 256 + c] = f2bf(mr > 0.f ? copysignf(mr, vr) : 0.f);
      Oi_[(long long)p * 256 + c] = f2bf(mi > 0.f ? copysignf(mi, vi) : 0.f);
    }
  }
}

// --------------------------------- driver ----------------------------------
extern "C" void kernel_launch(void* const* d_in, const int* in_sizes, int n_in,
                              void* d_out, int out_size, void* d_ws, size_t ws_size,
                              hipStream_t stream) {
  (void)in_sizes; (void)n_in; (void)out_size; (void)ws_size;
  const float* x   = (const float*)d_in[0];
  const float* n1g = (const float*)d_in[1];
  const float* n1b = (const float*)d_in[2];
  const float* n2g = (const float*)d_in[3];
  const float* n2b = (const float*)d_in[4];
  const float* w1  = (const float*)d_in[5];
  const float* b1  = (const float*)d_in[6];
  const float* w2  = (const float*)d_in[7];
  const float* b2  = (const float*)d_in[8];
  const float* mw1 = (const float*)d_in[9];
  const float* mb1 = (const float*)d_in[10];
  const float* mw2 = (const float*)d_in[11];
  const float* mb2 = (const float*)d_in[12];

  char* ws = (char*)d_ws;
  unsigned short* hln1 = (unsigned short*)(ws + OFF_HLN1);
  unsigned short* Xr   = (unsigned short*)(ws + OFF_XR);
  unsigned short* Xi   = (unsigned short*)(ws + OFF_XI);
  unsigned short* Yr   = (unsigned short*)(ws + OFF_YR);
  unsigned short* Yi   = (unsigned short*)(ws + OFF_YI);
  unsigned short* irf  = (unsigned short*)(ws + OFF_IRF);
  float*          out1 = (float*)(ws + OFF_OUT1);
  unsigned short* FwC  = (unsigned short*)(ws + OFF_MATS + MOFF_FWC);
  unsigned short* FwS  = (unsigned short*)(ws + OFF_MATS + MOFF_FWS);
  unsigned short* FhC  = (unsigned short*)(ws + OFF_MATS + MOFF_FHC);
  unsigned short* FhS  = (unsigned short*)(ws + OFF_MATS + MOFF_FHS);
  unsigned short* IFhC = (unsigned short*)(ws + OFF_MATS + MOFF_IFHC);
  unsigned short* IFhS = (unsigned short*)(ws + OFF_MATS + MOFF_IFHS);
  unsigned short* IFwC = (unsigned short*)(ws + OFF_MATS + MOFF_IFWC);
  unsigned short* IFwS = (unsigned short*)(ws + OFF_MATS + MOFF_IFWS);
  unsigned short* tB   = (unsigned short*)(ws + OFF_T);
  unsigned short* U    = (unsigned short*)(ws + OFF_U);
  const unsigned short* nullu = nullptr;
  const float* nullf = nullptr;

  build_mats_kernel<<<256, 256, 0, stream>>>(FwC, FwS, FhC, FhS, IFhC, IFhS, IFwC, IFwS);
  ln_kernel<<<NTOK, 256, 0, stream>>>(x, n1g, n1b, hln1);

  // W-direction rfft: per (b,h): X(129x256) = Fw(129x256) x hln1(256x256)
  dim3 g1(2, 2, 512);
  gemm_dual_kernel<unsigned short><<<g1, 256, 0, stream>>>(FwC, nullu, 256, 0,
      hln1, nullu, 256, 65536, 0u, Xr, 256, 33024, 129, 256, 256, 0, nullf, nullf);
  gemm_dual_kernel<unsigned short><<<g1, 256, 0, stream>>>(FwS, nullu, 256, 0,
      hln1, nullu, 256, 65536, 0u, Xi, 256, 33024, 129, 256, 256, 0, nullf, nullf);

  // H-direction complex FFT (fused dual): Yr = Fc*Xr - Fs*Xi ; Yi = Fc*Xi + Fs*Xr
  dim3 g2(258, 2, 2);
  gemm_dual_kernel<unsigned short><<<g2, 256, 0, stream>>>(FhC, FhS, 256, 0,
      Xr, Xi, 33024, 8454144, 0x8000u, Yr, 33024, 8454144,
      256, 33024, 256, 0, nullf, nullf);
  gemm_dual_kernel<unsigned short><<<g2, 256, 0, stream>>>(FhC, FhS, 256, 0,
      Xi, Xr, 33024, 8454144, 0u, Yi, 33024, 8454144,
      256, 33024, 256, 0, nullf, nullf);

  // spectral block MLP (writes filtered spectra into X planes)
  afno_filter_kernel<<<NPOS / 16, 256, 0, stream>>>(Yr, Yi, w1, b1, w2, b2, Xr, Xi);

  // inverse H FFT (fused dual): Zr = IC*Xr - IS*Xi ; Zi = IC*Xi + IS*Xr
  gemm_dual_kernel<unsigned short><<<g2, 256, 0, stream>>>(IFhC, IFhS, 256, 0,
      Xr, Xi, 33024, 8454144, 0x8000u, Yr, 33024, 8454144,
      256, 33024, 256, 0, nullf, nullf);
  gemm_dual_kernel<unsigned short><<<g2, 256, 0, stream>>>(IFhC, IFhS, 256, 0,
      Xi, Xr, 33024, 8454144, 0u, Yi, 33024, 8454144,
      256, 33024, 256, 0, nullf, nullf);

  // inverse W irfft (fused dual): irf = IFwC*Zr + IFwS*Zi, per (b,h)
  gemm_dual_kernel<unsigned short><<<g1, 256, 0, stream>>>(IFwC, IFwS, 129, 0,
      Yr, Yi, 256, 33024, 0u, irf, 256, 65536, 256, 256, 129, 0, nullf, nullf);

  // out1 = irfft + ln1 + x (double skip); t = LN2(out1)
  add_ln2_kernel<<<NTOK, 256, 0, stream>>>(irf, hln1, x, n2g, n2b, out1, tB);

  // token MLP: U = gelu(t @ mw1 + mb1) [bf16]; out = U @ mw2 + mb2 + out1
  gemm_dual_kernel<float><<<dim3(8, 1024, 1), 256, 0, stream>>>(tB, nullu, 256, 0,
      mw1, nullf, 1024, 0, 0u, U, 1024, 0, NTOK, 1024, 256, 1, mb1, nullf);
  gemm_dual_kernel<float><<<dim3(2, 1024, 1), 256, 0, stream>>>(U, nullu, 1024, 0,
      mw2, nullf, 256, 0, 0u, d_out, 256, 0, NTOK, 256, 1024, 2, mb2, out1);
}
